// ShotModel_GAT_52785148068161
// MI455X (gfx1250) — matrix operand order repaired and checked by reading, not compiled, not run on hardware
//
#include <hip/hip_runtime.h>

typedef __attribute__((ext_vector_type(16))) _Float16 v16h;
typedef __attribute__((ext_vector_type(8)))  _Float16 v8h;
typedef __attribute__((ext_vector_type(8)))  float    v8f;

#define WMMA_F32_F16(a, b, c) \
  __builtin_amdgcn_wmma_f32_16x16x32_f16(false, (a), false, (b), (short)0, (c), false, false)

__device__ __forceinline__ float sigmoid_f(float v) { return 1.f / (1.f + __expf(-v)); }
__device__ __forceinline__ float tanh_f(float v) {
  float e2 = __expf(2.f * v);
  return (e2 - 1.f) / (e2 + 1.f);
}

// ---------------------------------------------------------------------------
// WMMA fragment loaders (wave32, 16x16x32 f16)
//
// B (K=32 x N=16) in the contiguous k order used here (not the ISA table order): lanes 0-15 -> N=lane, K=0..15;
// lanes 16-31 -> N=lane-16, K=16..31. With weights stored row-major [3H][K]
// (exactly the PyTorch Wih/Whh layout, converted to f16), a lane reads
// W[n0+lane%16][k0 + (lane/16)*16 .. +15] contiguously (32B aligned load).
__device__ __forceinline__ v16h load_b_frag(const _Float16* __restrict__ WT, int ldk,
                                            int n0, int k0, int lane) {
  int n = n0 + (lane & 15);
  int k = k0 + ((lane >> 4) << 4);
  return *(const v16h*)(WT + n * ldk + k);
}

// A (M=16 x K=32) in the same contiguous k order as B: lane holds row m = m0+lane%16,
// elements 0..15 = K kb..kb+15, kb = (lane/16)*16.
__device__ __forceinline__ v16h load_a_frag(const _Float16* H, int ldk,
                                            int m0, int k0, int lane) {
  int m = m0 + (lane & 15);
  int kb = k0 + ((lane >> 4) << 4);
  const _Float16* p = H + m * ldk + kb;
  v8h lo = *(const v8h*)(p);
  v8h hi = *(const v8h*)(p + 8);
  v16h a;
#pragma unroll
  for (int i = 0; i < 8; ++i) { a[i] = lo[i]; a[8 + i] = hi[i]; }
  return a;
}

// C/D element e of v8f maps to (m = (lane/16)*8 + e, n = n0 + lane%16).

// ---------------------------------------------------------------------------
__global__ void f32_to_f16_kernel(const float* __restrict__ s, _Float16* __restrict__ d, int n) {
  int i = blockIdx.x * blockDim.x + threadIdx.x;
  if (i < n) d[i] = (_Float16)s[i];
}

__global__ void zero_kernel(float* __restrict__ p, int n) {
  int i = blockIdx.x * blockDim.x + threadIdx.x;
  if (i < n) p[i] = 0.f;
}

// ---------------------------------------------------------------------------
// na GRU: in=3, H=64. 64 nodes/block, 256 threads (8 waves).
// Wave w owns gate-column trio (w&3): columns {16c, 64+16c, 128+16c}, and
// M-tile pair (w>>2)*2. Per-wave: 6 v8f accumulators + 6 pinned B frags
// (~140 VGPRs) -> fits the 256-VGPR / 4-wave-per-SIMD budget without spills.
__global__ void __launch_bounds__(256, 4) na_gru_kernel(
    const float* __restrict__ x, const _Float16* __restrict__ whh16,
    const float* __restrict__ wih, const float* __restrict__ bih,
    const float* __restrict__ bhh, float* __restrict__ traj, int n_nodes)
{
  __shared__ _Float16 h16[64 * 64];
  __shared__ float    h32[64 * 64];
  __shared__ float    xs[64 * 3];
  const int tid = threadIdx.x;
  const int lane = tid & 31;
  const int wave = tid >> 5;
  const int nodeBase = blockIdx.x * 64;

  for (int i = tid; i < 64 * 64; i += 256) { h16[i] = (_Float16)0.f; h32[i] = 0.f; }

  const int j0 = (wave & 3) * 16;       // gate-column trio
  const int mtb = (wave >> 2) * 2;      // first of two owned M-tiles
  const int j = j0 + (lane & 15);
  // r/z gates: input and hidden biases always summed -> fold.
  const float bcr = bih[j] + bhh[j];
  const float bcz = bih[64 + j] + bhh[64 + j];
  const float bin_ = bih[128 + j];            // n gate: i_n bias
  const float bhn = bhh[128 + j];             // n gate: h_n bias (scaled by r)
  const float wr0 = wih[j * 3], wr1 = wih[j * 3 + 1], wr2 = wih[j * 3 + 2];
  const float wz0 = wih[(64 + j) * 3], wz1 = wih[(64 + j) * 3 + 1], wz2 = wih[(64 + j) * 3 + 2];
  const float wn0 = wih[(128 + j) * 3], wn1 = wih[(128 + j) * 3 + 1], wn2 = wih[(128 + j) * 3 + 2];

  v16h Br[2], Bz[2], Bn[2];
#pragma unroll
  for (int kt = 0; kt < 2; ++kt) {
    Br[kt] = load_b_frag(whh16, 64, j0, kt * 32, lane);
    Bz[kt] = load_b_frag(whh16, 64, 64 + j0, kt * 32, lane);
    Bn[kt] = load_b_frag(whh16, 64, 128 + j0, kt * 32, lane);
  }
  __syncthreads();

  const v8f zacc = {};
  for (int t = 0; t < 50; ++t) {
    for (int i = tid; i < 64 * 3; i += 256) {
      int node = i / 3, c = i - node * 3;
      int g = nodeBase + node; if (g > n_nodes - 1) g = n_nodes - 1;
      xs[i] = x[(g * 50 + t) * 3 + c];
    }
    __syncthreads();

    v8f aR[2], aZ[2], aN[2];
#pragma unroll
    for (int mt = 0; mt < 2; ++mt) { aR[mt] = zacc; aZ[mt] = zacc; aN[mt] = zacc; }
#pragma unroll
    for (int kt = 0; kt < 2; ++kt) {
#pragma unroll
      for (int mt = 0; mt < 2; ++mt) {
        v16h a = load_a_frag(h16, 64, (mtb + mt) * 16, kt * 32, lane);
        aR[mt] = WMMA_F32_F16(a, Br[kt], aR[mt]);
        aZ[mt] = WMMA_F32_F16(a, Bz[kt], aZ[mt]);
        aN[mt] = WMMA_F32_F16(a, Bn[kt], aN[mt]);
      }
    }
    __syncthreads();  // all WMMA A-reads of h16 complete before h updates

#pragma unroll
    for (int mt = 0; mt < 2; ++mt) {
#pragma unroll
      for (int e = 0; e < 8; ++e) {
        int m = (mtb + mt) * 16 + ((lane >> 4) << 3) + e;
        float x0 = xs[m * 3], x1 = xs[m * 3 + 1], x2 = xs[m * 3 + 2];
        float ir = fmaf(wr2, x2, fmaf(wr1, x1, fmaf(wr0, x0, bcr)));
        float iz = fmaf(wz2, x2, fmaf(wz1, x1, fmaf(wz0, x0, bcz)));
        float in_ = fmaf(wn2, x2, fmaf(wn1, x1, fmaf(wn0, x0, bin_)));
        float r = sigmoid_f(ir + aR[mt][e]);
        float z = sigmoid_f(iz + aZ[mt][e]);
        float nn = tanh_f(in_ + r * (aN[mt][e] + bhn));
        float hold = h32[m * 64 + j];
        float hnew = (1.f - z) * nn + z * hold;
        h32[m * 64 + j] = hnew;
        h16[m * 64 + j] = (_Float16)hnew;
      }
    }
    __syncthreads();
  }

  for (int i = tid; i < 64 * 64; i += 256) {
    int g = nodeBase + (i >> 6);
    if (g < n_nodes) traj[g * 64 + (i & 63)] = h32[i];
  }
}

// ---------------------------------------------------------------------------
// ta GRU, both layers fused: layer0 in=6 (scalar gi), layer1 in=128 (WMMA gi).
// 16 nodes/block (single M-tile), 256 threads (8 waves); wave w owns trio
// {16w, 128+16w, 256+16w}. Layer-1 r/z gates accumulate BOTH GEMMs into one
// v8f each -> per-wave accumulators are 4 v8f (layer1) / 3 v8f (layer0),
// total demand ~160 VGPRs: fits 256 cap without spills or MSB encoding.
// B frags streamed from global (L2-resident) each step; h0/h1 live in LDS.
__global__ void __launch_bounds__(256, 4) ta_gru_kernel(
    const float* __restrict__ x,
    const _Float16* __restrict__ whh0_16, const _Float16* __restrict__ wih1_16,
    const _Float16* __restrict__ whh1_16,
    const float* __restrict__ wih0,
    const float* __restrict__ bih0, const float* __restrict__ bhh0,
    const float* __restrict__ bih1, const float* __restrict__ bhh1,
    float* __restrict__ traj2, int n_nodes)
{
  __shared__ _Float16 h0_16[16 * 128];
  __shared__ float    h0_32[16 * 128];
  __shared__ _Float16 h1_16[16 * 128];
  __shared__ float    h1_32[16 * 128];
  __shared__ float    xs[16 * 6];
  const int tid = threadIdx.x;
  const int lane = tid & 31;
  const int wave = tid >> 5;
  const int nodeBase = blockIdx.x * 16;

  for (int i = tid; i < 16 * 128; i += 256) {
    h0_16[i] = (_Float16)0.f; h0_32[i] = 0.f;
    h1_16[i] = (_Float16)0.f; h1_32[i] = 0.f;
  }

  const int j0 = wave * 16;
  const int j = j0 + (lane & 15);
  const float b0r = bih0[j] + bhh0[j];
  const float b0z = bih0[128 + j] + bhh0[128 + j];
  const float b0in = bih0[256 + j];
  const float b0hn = bhh0[256 + j];
  const float b1r = bih1[j] + bhh1[j];
  const float b1z = bih1[128 + j] + bhh1[128 + j];
  const float b1in = bih1[256 + j];
  const float b1hn = bhh1[256 + j];
  float w0r[6], w0z[6], w0n[6];
#pragma unroll
  for (int c = 0; c < 6; ++c) {
    w0r[c] = wih0[j * 6 + c];
    w0z[c] = wih0[(128 + j) * 6 + c];
    w0n[c] = wih0[(256 + j) * 6 + c];
  }
  __syncthreads();

  const v8f zacc = {};
  for (int t = 0; t < 50; ++t) {
    for (int i = tid; i < 16 * 6; i += 256) {
      int node = i / 6, c = i - node * 6;
      int g = nodeBase + node; if (g > n_nodes - 1) g = n_nodes - 1;
      xs[i] = (c < 3) ? x[t * 3 + c]                     // ball = node 0
                      : x[(g * 50 + t) * 3 + (c - 3)];   // own trajectory
    }
    __syncthreads();

    // ---- layer 0: gh0 = h0 @ Whh0^T ----
    v8f aR = zacc, aZ = zacc, aN = zacc;
#pragma unroll
    for (int kt = 0; kt < 4; ++kt) {
      v16h br = load_b_frag(whh0_16, 128, j0, kt * 32, lane);
      v16h bz = load_b_frag(whh0_16, 128, 128 + j0, kt * 32, lane);
      v16h bn = load_b_frag(whh0_16, 128, 256 + j0, kt * 32, lane);
      v16h a = load_a_frag(h0_16, 128, 0, kt * 32, lane);
      aR = WMMA_F32_F16(a, br, aR);
      aZ = WMMA_F32_F16(a, bz, aZ);
      aN = WMMA_F32_F16(a, bn, aN);
    }
    __syncthreads();
#pragma unroll
    for (int e = 0; e < 8; ++e) {
      int m = ((lane >> 4) << 3) + e;
      float ir = b0r, iz = b0z, in_ = b0in;
#pragma unroll
      for (int c = 0; c < 6; ++c) {
        float xv = xs[m * 6 + c];
        ir = fmaf(w0r[c], xv, ir);
        iz = fmaf(w0z[c], xv, iz);
        in_ = fmaf(w0n[c], xv, in_);
      }
      float r = sigmoid_f(ir + aR[e]);
      float z = sigmoid_f(iz + aZ[e]);
      float nn = tanh_f(in_ + r * (aN[e] + b0hn));
      float hold = h0_32[m * 128 + j];
      float hnew = (1.f - z) * nn + z * hold;
      h0_32[m * 128 + j] = hnew;
      h0_16[m * 128 + j] = (_Float16)hnew;
    }
    __syncthreads();  // h0_new visible to all waves

    // ---- layer 1: r/z accumulate gi1+gh1 together; n keeps i_n / h_n split ----
    v8f cR = zacc, cZ = zacc, iN = zacc, hN = zacc;
#pragma unroll
    for (int kt = 0; kt < 4; ++kt) {
      {
        v16h br = load_b_frag(wih1_16, 128, j0, kt * 32, lane);
        v16h bz = load_b_frag(wih1_16, 128, 128 + j0, kt * 32, lane);
        v16h bn = load_b_frag(wih1_16, 128, 256 + j0, kt * 32, lane);
        v16h a = load_a_frag(h0_16, 128, 0, kt * 32, lane);
        cR = WMMA_F32_F16(a, br, cR);
        cZ = WMMA_F32_F16(a, bz, cZ);
        iN = WMMA_F32_F16(a, bn, iN);
      }
      {
        v16h br = load_b_frag(whh1_16, 128, j0, kt * 32, lane);
        v16h bz = load_b_frag(whh1_16, 128, 128 + j0, kt * 32, lane);
        v16h bn = load_b_frag(whh1_16, 128, 256 + j0, kt * 32, lane);
        v16h a = load_a_frag(h1_16, 128, 0, kt * 32, lane);
        cR = WMMA_F32_F16(a, br, cR);
        cZ = WMMA_F32_F16(a, bz, cZ);
        hN = WMMA_F32_F16(a, bn, hN);
      }
    }
    __syncthreads();
#pragma unroll
    for (int e = 0; e < 8; ++e) {
      int m = ((lane >> 4) << 3) + e;
      float r = sigmoid_f(cR[e] + b1r);
      float z = sigmoid_f(cZ[e] + b1z);
      float nn = tanh_f(iN[e] + b1in + r * (hN[e] + b1hn));
      float hold = h1_32[m * 128 + j];
      float hnew = (1.f - z) * nn + z * hold;
      h1_32[m * 128 + j] = hnew;
      h1_16[m * 128 + j] = (_Float16)hnew;
    }
    __syncthreads();
  }

  for (int i = tid; i < 16 * 128; i += 256) {
    int g = nodeBase + (i >> 7);
    if (g < n_nodes) traj2[g * 128 + (i & 127)] = h1_32[i];
  }
}

// ---------------------------------------------------------------------------
// GAT node transform: h = feat @ Wfc^T (64x64, WMMA), plus attention dots
// a_src = h . Watt[:64], a_dst = h . Watt[64:]. Optional ELU on input.
__global__ void __launch_bounds__(128) gat_transform_kernel(
    const float* __restrict__ feat, const _Float16* __restrict__ wfc16,
    const float* __restrict__ watt,
    float* __restrict__ hout, float* __restrict__ a_src, float* __restrict__ a_dst,
    int n_nodes, int apply_elu)
{
  __shared__ _Float16 fs[64 * 64];
  __shared__ float    hs[64 * 64];
  const int tid = threadIdx.x;
  const int lane = tid & 31;
  const int wave = tid >> 5;
  const int nodeBase = blockIdx.x * 64;

  for (int i = tid; i < 64 * 64; i += 128) {
    int g = nodeBase + (i >> 6); if (g > n_nodes - 1) g = n_nodes - 1;
    float v = feat[g * 64 + (i & 63)];
    if (apply_elu) v = (v > 0.f) ? v : (__expf(v) - 1.f);
    fs[i] = (_Float16)v;
  }
  __syncthreads();

  const int n0 = wave * 16;
  const v8f zacc = {};
  v8f acc[4];
#pragma unroll
  for (int mt = 0; mt < 4; ++mt) acc[mt] = zacc;
#pragma unroll
  for (int kt = 0; kt < 2; ++kt) {
    v16h b = load_b_frag(wfc16, 64, n0, kt * 32, lane);
#pragma unroll
    for (int mt = 0; mt < 4; ++mt) {
      v16h a = load_a_frag(fs, 64, mt * 16, kt * 32, lane);
      acc[mt] = WMMA_F32_F16(a, b, acc[mt]);
    }
  }
  const int n = n0 + (lane & 15);
#pragma unroll
  for (int mt = 0; mt < 4; ++mt)
#pragma unroll
    for (int e = 0; e < 8; ++e) {
      int m = mt * 16 + ((lane >> 4) << 3) + e;
      hs[m * 64 + n] = acc[mt][e];
    }
  __syncthreads();

  if (tid < 64) {
    float s = 0.f, d = 0.f;
#pragma unroll 8
    for (int c = 0; c < 64; ++c) {
      float hv = hs[tid * 64 + c];
      s = fmaf(hv, watt[c], s);
      d = fmaf(hv, watt[64 + c], d);
    }
    int g = nodeBase + tid;
    if (g < n_nodes) { a_src[g] = s; a_dst[g] = d; }
  }
  for (int i = tid; i < 64 * 64; i += 128) {
    int g = nodeBase + (i >> 6);
    if (g < n_nodes) hout[g * 64 + (i & 63)] = hs[i];
  }
}

// One wave per edge; lane handles 2 channels. out must be pre-zeroed.
__global__ void __launch_bounds__(256) gat_aggregate_kernel(
    const int* __restrict__ src, const int* __restrict__ dst,
    const float* __restrict__ a_src, const float* __restrict__ a_dst,
    const float* __restrict__ h, float* __restrict__ out, int n_edges)
{
  int waveId = (blockIdx.x * blockDim.x + threadIdx.x) >> 5;
  int lane = threadIdx.x & 31;
  if (waveId >= n_edges) return;
  int s = src[waveId], d = dst[waveId];
  float e = sigmoid_f(a_src[s] + a_dst[d]);
  int c = lane * 2;
  atomicAdd(&out[d * 64 + c],     e * h[s * 64 + c]);
  atomicAdd(&out[d * 64 + c + 1], e * h[s * 64 + c + 1]);
}

// ---------------------------------------------------------------------------
// Predictor: gather [gfeat|tf2] -> 192, hid = relu(@W1^T + b1) via WMMA,
// logit = hid.W2 + b2, sigmoid. 64 queries/block, 128 threads.
__global__ void __launch_bounds__(128) predictor_kernel(
    const int* __restrict__ qf, const float* __restrict__ gfeat,
    const float* __restrict__ tf2, const _Float16* __restrict__ w1_16,
    const float* __restrict__ b1, const float* __restrict__ w2,
    const float* __restrict__ b2, float* __restrict__ out, int n_q)
{
  __shared__ _Float16 us[64 * 192];
  __shared__ float    hid[64 * 128];
  const int tid = threadIdx.x;
  const int lane = tid & 31;
  const int wave = tid >> 5;
  const int base = blockIdx.x * 64;

  for (int i = tid; i < 64 * 192; i += 128) {
    int node = i / 192, c = i - node * 192;
    int qi = base + node; if (qi > n_q - 1) qi = n_q - 1;
    int g = qf[qi];
    float v = (c < 64) ? gfeat[g * 64 + c] : tf2[g * 128 + (c - 64)];
    us[i] = (_Float16)v;
  }
  __syncthreads();

  const v8f zacc = {};
#pragma unroll
  for (int ni = 0; ni < 2; ++ni) {
    int n0 = (wave * 2 + ni) * 16;
    v8f acc[4];
#pragma unroll
    for (int mt = 0; mt < 4; ++mt) acc[mt] = zacc;
#pragma unroll
    for (int kt = 0; kt < 6; ++kt) {
      v16h b = load_b_frag(w1_16, 192, n0, kt * 32, lane);
#pragma unroll
      for (int mt = 0; mt < 4; ++mt) {
        v16h a = load_a_frag(us, 192, mt * 16, kt * 32, lane);
        acc[mt] = WMMA_F32_F16(a, b, acc[mt]);
      }
    }
    int n = n0 + (lane & 15);
    float bias = b1[n];
#pragma unroll
    for (int mt = 0; mt < 4; ++mt)
#pragma unroll
      for (int e = 0; e < 8; ++e) {
        int m = mt * 16 + ((lane >> 4) << 3) + e;
        float v = acc[mt][e] + bias;
        hid[m * 128 + n] = (v > 0.f) ? v : 0.f;
      }
  }
  __syncthreads();

  if (tid < 64) {
    float s = b2[0];
#pragma unroll 8
    for (int c = 0; c < 128; ++c) s = fmaf(hid[tid * 128 + c], w2[c], s);
    int qi = base + tid;
    if (qi < n_q) out[qi] = sigmoid_f(s);
  }
}

// ---------------------------------------------------------------------------
extern "C" void kernel_launch(void* const* d_in, const int* in_sizes, int n_in,
                              void* d_out, int out_size, void* d_ws, size_t ws_size,
                              hipStream_t stream) {
  const float* x      = (const float*)d_in[0];
  const int*   src    = (const int*)d_in[1];
  const int*   dst    = (const int*)d_in[2];
  const int*   qf     = (const int*)d_in[3];
  const float* naWih  = (const float*)d_in[4];
  const float* naWhh  = (const float*)d_in[5];
  const float* naBih  = (const float*)d_in[6];
  const float* naBhh  = (const float*)d_in[7];
  const float* taWih0 = (const float*)d_in[8];
  const float* taWhh0 = (const float*)d_in[9];
  const float* taBih0 = (const float*)d_in[10];
  const float* taBhh0 = (const float*)d_in[11];
  const float* taWih1 = (const float*)d_in[12];
  const float* taWhh1 = (const float*)d_in[13];
  const float* taBih1 = (const float*)d_in[14];
  const float* taBhh1 = (const float*)d_in[15];
  const float* g1Wfc  = (const float*)d_in[16];
  const float* g1Watt = (const float*)d_in[17];
  const float* g2Wfc  = (const float*)d_in[18];
  const float* g2Watt = (const float*)d_in[19];
  const float* pW1    = (const float*)d_in[20];
  const float* pB1    = (const float*)d_in[21];
  const float* pW2    = (const float*)d_in[22];
  const float* pB2    = (const float*)d_in[23];

  const int n_nodes = in_sizes[0] / (50 * 3);
  const int n_edges = in_sizes[1];
  const int n_q     = in_sizes[3];

  // Workspace carve-up (256B aligned regions)
  char* base = (char*)d_ws;
  size_t off = 0;
  auto alloc = [&](size_t bytes) -> char* {
    off = (off + 255) & ~(size_t)255;
    char* p = base + off;
    off += bytes;
    return p;
  };
  _Float16* whhNa = (_Float16*)alloc(192 * 64 * 2);
  _Float16* whh0  = (_Float16*)alloc(384 * 128 * 2);
  _Float16* wih1  = (_Float16*)alloc(384 * 128 * 2);
  _Float16* whh1  = (_Float16*)alloc(384 * 128 * 2);
  _Float16* wfc1  = (_Float16*)alloc(64 * 64 * 2);
  _Float16* wfc2  = (_Float16*)alloc(64 * 64 * 2);
  _Float16* w1p   = (_Float16*)alloc(128 * 192 * 2);
  float* traj   = (float*)alloc((size_t)n_nodes * 64 * 4);
  float* traj2  = (float*)alloc((size_t)n_nodes * 128 * 4);
  float* h1buf  = (float*)alloc((size_t)n_nodes * 64 * 4);
  float* as1    = (float*)alloc((size_t)n_nodes * 4);
  float* ad1    = (float*)alloc((size_t)n_nodes * 4);
  float* agg1   = (float*)alloc((size_t)n_nodes * 64 * 4);
  float* h2buf  = (float*)alloc((size_t)n_nodes * 64 * 4);
  float* as2    = (float*)alloc((size_t)n_nodes * 4);
  float* ad2    = (float*)alloc((size_t)n_nodes * 4);
  float* gfeat  = (float*)alloc((size_t)n_nodes * 64 * 4);

  auto cvt = [&](const float* s, _Float16* d, int n) {
    f32_to_f16_kernel<<<(n + 255) / 256, 256, 0, stream>>>(s, d, n);
  };
  cvt(naWhh,  whhNa, 192 * 64);
  cvt(taWhh0, whh0,  384 * 128);
  cvt(taWih1, wih1,  384 * 128);
  cvt(taWhh1, whh1,  384 * 128);
  cvt(g1Wfc,  wfc1,  64 * 64);
  cvt(g2Wfc,  wfc2,  64 * 64);
  cvt(pW1,    w1p,   128 * 192);

  // GRUs
  na_gru_kernel<<<(n_nodes + 63) / 64, 256, 0, stream>>>(
      x, whhNa, naWih, naBih, naBhh, traj, n_nodes);
  ta_gru_kernel<<<(n_nodes + 15) / 16, 256, 0, stream>>>(
      x, whh0, wih1, whh1, taWih0, taBih0, taBhh0, taBih1, taBhh1, traj2, n_nodes);

  // GAT layer 1
  gat_transform_kernel<<<(n_nodes + 63) / 64, 128, 0, stream>>>(
      traj, wfc1, g1Watt, h1buf, as1, ad1, n_nodes, 0);
  zero_kernel<<<((n_nodes * 64) + 255) / 256, 256, 0, stream>>>(agg1, n_nodes * 64);
  gat_aggregate_kernel<<<(n_edges + 7) / 8, 256, 0, stream>>>(
      src, dst, as1, ad1, h1buf, agg1, n_edges);

  // GAT layer 2 (ELU applied to layer-1 aggregate on load)
  gat_transform_kernel<<<(n_nodes + 63) / 64, 128, 0, stream>>>(
      agg1, wfc2, g2Watt, h2buf, as2, ad2, n_nodes, 1);
  zero_kernel<<<((n_nodes * 64) + 255) / 256, 256, 0, stream>>>(gfeat, n_nodes * 64);
  gat_aggregate_kernel<<<(n_edges + 7) / 8, 256, 0, stream>>>(
      src, dst, as2, ad2, h2buf, gfeat, n_edges);

  // Predictor
  predictor_kernel<<<(n_q + 63) / 64, 128, 0, stream>>>(
      qf, gfeat, traj2, w1p, pB1, pW2, pB2, (float*)d_out, n_q);
}
